// DynamicConv2D_38096359915591
// MI455X (gfx1250) — matrix operand order repaired and checked by reading, not compiled, not run on hardware
//
#include <hip/hip_runtime.h>
#include <hip/hip_bf16.h>
#include <math.h>

#define B_     32
#define CIN_   64
#define COUT_  64
#define K_     3
#define ADIM_  64
#define H_     128
#define W_     128
#define HW_    (H_ * W_)
#define WCOLS_ 36864          /* COUT*CIN*K*K */

typedef __attribute__((ext_vector_type(16))) __bf16 v16bf;
typedef __attribute__((ext_vector_type(8)))  __bf16 v8bf;
typedef __attribute__((ext_vector_type(8)))  float  v8f;

static __device__ __forceinline__ __bf16 f32_to_bf16_rne(float f) {
  unsigned u = __float_as_uint(f);
  u += 0x7FFFu + ((u >> 16) & 1u);          // round to nearest even
  unsigned short s = (unsigned short)(u >> 16);
  return __builtin_bit_cast(__bf16, s);
}

// ---------------- Kernel 1: global average pool  pooled[b,cin] ----------------
__global__ void dyc_pool_kernel(const float* __restrict__ x, float* __restrict__ pooled) {
  __shared__ float red[256];
  const int bc = blockIdx.x;                       // b*CIN + cin
  const float* p = x + (size_t)bc * HW_;
  float s = 0.f;
  for (int i = threadIdx.x; i < HW_; i += 256) s += p[i];
  red[threadIdx.x] = s;
  __syncthreads();
  for (int off = 128; off > 0; off >>= 1) {
    if (threadIdx.x < off) red[threadIdx.x] += red[threadIdx.x + off];
    __syncthreads();
  }
  if (threadIdx.x == 0) pooled[bc] = red[0] * (1.0f / HW_);
}

// ---------------- Kernel 2: h = relu(pooled @ w1 + b1) ----------------
__global__ void dyc_mlp1_kernel(const float* __restrict__ pooled,
                                const float* __restrict__ w1,
                                const float* __restrict__ b1,
                                float* __restrict__ hbuf) {
  __shared__ float pr[CIN_];
  const int b = blockIdx.x, j = threadIdx.x;       // 64 threads
  pr[j] = pooled[b * CIN_ + j];
  __syncthreads();
  float s = b1[j];
  #pragma unroll 8
  for (int c = 0; c < CIN_; ++c) s += pr[c] * w1[c * ADIM_ + j];
  hbuf[b * ADIM_ + j] = fmaxf(s, 0.f);
}

// ---- Kernel 3: adapted = base_w + scale*tanh(h@w2+b2), stored bf16 as aw[b][tap][cout][cin]
__global__ void dyc_adapt_kernel(const float* __restrict__ hbuf,
                                 const float* __restrict__ w2,
                                 const float* __restrict__ b2,
                                 const float* __restrict__ base_w,
                                 const float* __restrict__ scale_p,
                                 __bf16* __restrict__ aw) {
  __shared__ float hr[ADIM_];
  const int b   = blockIdx.x / 144;                // 144 blocks of 256 per sample
  const int col = (blockIdx.x % 144) * 256 + threadIdx.x;
  if (threadIdx.x < ADIM_) hr[threadIdx.x] = hbuf[b * ADIM_ + threadIdx.x];
  __syncthreads();
  float s = b2[col];
  #pragma unroll 8
  for (int a = 0; a < ADIM_; ++a) s += hr[a] * w2[(size_t)a * WCOLS_ + col];
  const float val = base_w[col] + scale_p[0] * tanhf(s);
  const int cout = col / 576;
  const int rem  = col % 576;
  const int cin  = rem / 9;
  const int tap  = rem % 9;                        // ky*3+kx
  aw[(((size_t)b * 9 + tap) * COUT_ + cout) * CIN_ + cin] = f32_to_bf16_rne(val);
}

// ---------------- Kernel 4: implicit-GEMM conv via bf16 WMMA ----------------
// Grid: B*H*(W/64) blocks of 128 threads (4 waves).
// Each WG: one sample b, one output row y, one 64-pixel strip.
// GEMM: D[64cout, 64pix] += sum over 9 taps of  Wtap[64,64cin] x Xshift[64cin, 64pix].
#define LCIN 72                                    // padded cin stride (16B aligned, no bank conflicts)
#define LCOL 66
__global__ __launch_bounds__(128) void dyc_conv_kernel(const float* __restrict__ x,
                                                       const __bf16* __restrict__ aw,
                                                       const float* __restrict__ base_b,
                                                       float* __restrict__ out) {
  __shared__ __bf16 xs[3 * LCOL * LCIN];

  const int blk = blockIdx.x;
  const int b   = blk >> 8;                        // 256 blocks per sample
  const int rem = blk & 255;
  const int y   = rem >> 1;
  const int xb  = (rem & 1) << 6;

  // ---- stage 3x66x64 input window into LDS, bf16, cin innermost (stride LCIN) ----
  for (int idx = threadIdx.x; idx < 3 * LCOL * CIN_; idx += 128) {   // exactly 99 iters/thread
    const int c   = idx % LCOL;
    const int t   = idx / LCOL;
    const int r   = t % 3;
    const int cin = t / 3;
    const int xx  = xb - 1 + c;
    const int yy  = y - 1 + r;
    float v = 0.f;
    if (xx >= 0 && xx < W_ && yy >= 0 && yy < H_)
      v = x[(((size_t)b * CIN_ + cin) * H_ + yy) * W_ + xx];
    xs[(r * LCOL + c) * LCIN + cin] = f32_to_bf16_rne(v);
  }
  __syncthreads();

  const int lane = threadIdx.x & 31;
  const int wave = threadIdx.x >> 5;
  const int lr   = lane & 15;
  const int hi   = lane >> 4;
  const int m0   = wave << 4;                      // cout tile base per wave
  const int hi8  = hi << 3;

  // accumulators: 4 N-tiles of 16 pixels; bias folded in
  v8f acc[4];
  {
    v8f ci;
    #pragma unroll
    for (int v = 0; v < 8; ++v) ci[v] = base_b[m0 + hi8 + v];
    acc[0] = ci; acc[1] = ci; acc[2] = ci; acc[3] = ci;
  }

  const __bf16* awbase = aw + (((size_t)b * 9) * COUT_ + (m0 + lr)) * CIN_;

  for (int tap = 0; tap < 9; ++tap) {
    const int ky = tap / 3;
    const int kx = tap - ky * 3;
    const __bf16* arow = awbase + (size_t)tap * (COUT_ * CIN_);
    __builtin_prefetch(arow + COUT_ * CIN_, 0, 1);         // next tap's weight slab

    #pragma unroll
    for (int kh = 0; kh < 2; ++kh) {                       // two K=32 chunks of cin
      // A fragment: row cout=m0+lr; 16-bit A lane layout: K = {hi*8+0..7, 16+hi*8+0..7}
      const __bf16* ap = arow + kh * 32 + hi8;
      const v8bf alo = *(const v8bf*)ap;
      const v8bf ahi = *(const v8bf*)(ap + 16);
      const v16bf av = __builtin_shufflevector(alo, ahi,
          0,1,2,3,4,5,6,7,8,9,10,11,12,13,14,15);

      #pragma unroll
      for (int nt = 0; nt < 4; ++nt) {
        // B fragment: col pixel nt*16+lr; 16-bit B lane layout: K = {hi*8+0..7, 16+hi*8+0..7}
        const int c = (nt << 4) + lr + kx;                 // shifted lds column
        const __bf16* bp = &xs[(ky * LCOL + c) * LCIN + kh * 32 + hi8];
        const v8bf blo = *(const v8bf*)bp;
        const v8bf bhi = *(const v8bf*)(bp + 16);
        const v16bf bv = __builtin_shufflevector(blo, bhi,
            0,1,2,3,4,5,6,7,8,9,10,11,12,13,14,15);

        acc[nt] = __builtin_amdgcn_wmma_f32_16x16x32_bf16(
            /*neg_a=*/false, av, /*neg_b=*/false, bv,
            /*c_mod=*/(short)0, acc[nt], /*reuse_a=*/false, /*reuse_b=*/false);
      }
    }
  }

  // ---- store D: lane holds pixel n=lr per tile, VGPR v holds cout = m0+hi*8+v ----
  #pragma unroll
  for (int v = 0; v < 8; ++v) {
    const int cout = m0 + hi8 + v;
    float* op = out + (((size_t)b * COUT_ + cout) * H_ + y) * W_ + xb + lr;
    op[0]  = acc[0][v];
    op[16] = acc[1][v];
    op[32] = acc[2][v];
    op[48] = acc[3][v];
  }
}

// ---------------- launcher ----------------
extern "C" void kernel_launch(void* const* d_in, const int* in_sizes, int n_in,
                              void* d_out, int out_size, void* d_ws, size_t ws_size,
                              hipStream_t stream) {
  const float* x      = (const float*)d_in[0];
  const float* base_w = (const float*)d_in[1];
  const float* base_b = (const float*)d_in[2];
  const float* w1     = (const float*)d_in[3];
  const float* b1     = (const float*)d_in[4];
  const float* w2     = (const float*)d_in[5];
  const float* b2     = (const float*)d_in[6];
  const float* scale  = (const float*)d_in[7];
  float* out = (float*)d_out;

  // workspace: pooled[32*64] f32 | hbuf[32*64] f32 | aw bf16[32*9*64*64]
  float*  pooled = (float*)d_ws;
  float*  hbuf   = (float*)((char*)d_ws + 8192);
  __bf16* aw     = (__bf16*)((char*)d_ws + 16384);

  dyc_pool_kernel <<<B_ * CIN_,   256, 0, stream>>>(x, pooled);
  dyc_mlp1_kernel <<<B_,           64, 0, stream>>>(pooled, w1, b1, hbuf);
  dyc_adapt_kernel<<<B_ * 144,    256, 0, stream>>>(hbuf, w2, b2, base_w, scale, aw);
  dyc_conv_kernel <<<B_ * H_ * (W_ / 64), 128, 0, stream>>>(x, aw, base_b, out);
}
